// FCOS_59974923321927
// MI455X (gfx1250) — matrix operand order repaired and checked by reading, not compiled, not run on hardware
//
#include <hip/hip_runtime.h>

// ---------------------------------------------------------------------------
// FCOS forward for MI455X (gfx1250). All conv layers run as implicit GEMM on
// V_WMMA_F32_16X16X32_F16. Activations are NHWC f16 and the GEMM K-dimension
// is ordered filter-tap-major (k = (kh*ks+kw)*Cin + cin), so for every conv
// with Cin%32==0 the per-lane B fragment is ONE aligned 32-byte vector load
// (B 32x16 wave32 layout), the A fragment is one 32-byte load from fragment-
// packed weights (A 16x32 layout), and the D store is one 16-byte v8h store
// (NHWC channel-contiguous). Each wave computes MBLK*16 x 16 outputs; one B
// fragment feeds MBLK WMMAs. conv1 (Cin=3) uses a small gather variant.
// Decode / top-k (LDS bitonic) / NMS (LDS greedy) are scalar epilogue kernels.
// ---------------------------------------------------------------------------

typedef __attribute__((ext_vector_type(16))) _Float16 v16h;
typedef __attribute__((ext_vector_type(8)))  _Float16 v8h;
typedef __attribute__((ext_vector_type(8)))  float    v8f;

#define IMG   512
#define NCLS  20
#define FPNC  128
#define TOPKN 1000
#define NLOC  5376          // 64*64 + 32*32 + 16*16
#define NPAD  8192          // next pow2 >= NLOC
#define NB    8

__device__ __forceinline__ float sigmoidf_(float x) { return 1.f / (1.f + expf(-x)); }

// ---------------- NCHW f32 -> NHWC f16 -------------------------------------
__global__ void k_nchw_to_nhwc(const float* __restrict__ src, _Float16* __restrict__ dst,
                               int B, int C, int H, int W) {
  int i = blockIdx.x * blockDim.x + threadIdx.x;   // NHWC linear
  int tot = B * C * H * W;
  if (i >= tot) return;
  int c = i % C; int p = i / C;
  int x = p % W; int t = p / W; int y = t % H; int b = t / H;
  dst[i] = (_Float16)src[(((size_t)b * C + c) * H + y) * W + x];
}

// OIHW f32 -> fragment-ordered f16, tap-major K. Element i decomposes as
//   t = i & 15, lane = (i>>4) & 31, blk = i>>9, mt = blk/ktiles, kt = blk%ktiles
//   m = mt*16 + (lane&15)
//   k = kt*32 + (lane>>4)*16 + t
// with k -> tap fp = k/Cin, channel c = k%Cin, kh = fp/ks, kw = fp%ks.
// Contiguous k order (k = 16*half + t), the same order the B load in k_conv_wmma uses.
__global__ void k_pack_w(const float* __restrict__ w, _Float16* __restrict__ out,
                         int Cout, int Cin, int ks, int K, int Kp, int Mp) {
  int i = blockIdx.x * blockDim.x + threadIdx.x;
  int tot = Mp * Kp;
  if (i >= tot) return;
  int ktiles = Kp >> 5;
  int t    = i & 15;
  int lane = (i >> 4) & 31;
  int blk  = i >> 9;
  int mt = blk / ktiles, kt = blk - mt * ktiles;
  int half = lane >> 4, l = lane & 15;
  int m = mt * 16 + l;
  int k = kt * 32 + half * 16 + t;
  float v = 0.f;
  if (m < Cout && k < K) {
    int fp = k / Cin, c = k - fp * Cin;
    int kh = fp / ks, kw = fp - kh * ks;
    v = w[(((size_t)m * Cin + c) * ks + kh) * ks + kw];
  }
  out[i] = (_Float16)v;
}

// ---------------- implicit-GEMM conv via WMMA (vectorized B) ---------------
// Requires Cin % 32 == 0 and Cin/32 a power of two (cshift = log2(Cin/32)).
template <int MBLK, int KS, bool RELU, bool OUTF32>
__global__ __launch_bounds__(32)
void k_conv_wmma(const _Float16* __restrict__ X, const _Float16* __restrict__ Wp,
                 const float* __restrict__ bias, void* __restrict__ Y,
                 int B, int Cin, int Hin, int Win,
                 int Cout, int Hout, int Wout,
                 int stride, int pad, int K, int cshift) {
  const int lane = threadIdx.x;
  const int half = lane >> 4;     // 0: lanes 0-15, 1: lanes 16-31
  const int l    = lane & 15;
  const int tileN = blockIdx.x, tileM = blockIdx.y;
  const int Ntot = B * Hout * Wout;

  const int n = tileN * 16 + l;   // output pixel (B*H*W index)
  int ox = 0, oy = 0, bi = 0;
  const bool nvalid = (n < Ntot);
  if (nvalid) { ox = n % Wout; int t = n / Wout; oy = t % Hout; bi = t / Hout; }

  const int ktiles = K >> 5;
  const v16h* __restrict__ Af = (const v16h*)Wp;
  const size_t abase = (size_t)(tileM * MBLK) * ktiles;
  const int cmask = (1 << cshift) - 1;

  v8f acc[MBLK] = {};
  v16h bz = {};

  for (int kt = 0; kt < ktiles; ++kt) {
    // prefetch next k-chunk of A fragments (global_prefetch_b8)
    if (kt + 1 < ktiles)
      __builtin_prefetch((const void*)(Af + (abase + kt + 1) * 32 + lane), 0, 3);

    // tap (kh,kw) and channel block for this chunk (shift/mask: Cin/32 is pow2)
    int kh = 0, kw = 0, cb;
    if (KS == 3) { int fp = kt >> cshift; kh = fp / 3; kw = fp - kh * 3; cb = kt & cmask; }
    else         { cb = kt; }
    const int iy = oy * stride + kh - pad;
    const int ix = ox * stride + kw - pad;

    // ---- B fragment: ONE aligned 32B vector load (16 contiguous channels) --
    v16h bf = bz;
    if (nvalid && (unsigned)iy < (unsigned)Hin && (unsigned)ix < (unsigned)Win) {
      const _Float16* px = X + ((((size_t)bi * Hin + iy) * Win + ix) * Cin + (cb << 5));
      bf = *(const v16h*)(px + (half << 4));
    }

    // ---- MBLK WMMAs share one B fragment; A is one 32B vector load each ----
#pragma unroll
    for (int mb = 0; mb < MBLK; ++mb) {
      v16h a = Af[(abase + (size_t)mb * ktiles + kt) * 32 + lane];
      acc[mb] = __builtin_amdgcn_wmma_f32_16x16x32_f16(false, a, false, bf,
                                                       (short)0, acc[mb], false, false);
    }
  }

  if (!nvalid) return;
  const size_t base = (((size_t)bi * Hout + oy) * Wout + ox) * Cout;   // NHWC
#pragma unroll
  for (int mb = 0; mb < MBLK; ++mb) {
    const int mbase = (tileM * MBLK + mb) * 16 + half * 8;
    if (!OUTF32 && mbase + 8 <= Cout) {
      v8h o;
#pragma unroll
      for (int r = 0; r < 8; ++r) {
        float v = acc[mb][r] + bias[mbase + r];
        if (RELU) v = fmaxf(v, 0.f);
        o[r] = (_Float16)v;
      }
      *(v8h*)((_Float16*)Y + base + mbase) = o;     // 16B NHWC store
    } else {
#pragma unroll
      for (int r = 0; r < 8; ++r) {
        int m = mbase + r;
        if (m < Cout) {
          float v = acc[mb][r] + bias[m];
          if (RELU) v = fmaxf(v, 0.f);
          if (OUTF32) ((float*)Y)[base + m] = v;
          else        ((_Float16*)Y)[base + m] = (_Float16)v;
        }
      }
    }
  }
}

// ---------------- conv1: Cin=3, K=27 (one k-chunk), gather B ---------------
__global__ __launch_bounds__(32)
void k_conv1_wmma(const _Float16* __restrict__ X, const _Float16* __restrict__ Wp,
                  const float* __restrict__ bias, _Float16* __restrict__ Y) {
  const int lane = threadIdx.x, half = lane >> 4, l = lane & 15;
  const int n = blockIdx.x * 16 + l;            // NB*256*256 is a multiple of 16
  const int ox = n & 255; int t = n >> 8; const int oy = t & 255; const int bi = t >> 8;

  v16h bf = {};
#pragma unroll
  for (int t2 = 0; t2 < 16; ++t2) {
    int k = half * 16 + t2;                     // tap-major: k = fp*3 + c
    if (k < 27) {
      int fp = k / 3, c = k - fp * 3;
      int kh = fp / 3, kw = fp - kh * 3;
      int iy = oy * 2 + kh, ix = ox * 2 + kw;   // stride 2, pad 0
      if (iy < 512 && ix < 512)
        bf[t2] = X[(((size_t)bi * 512 + iy) * 512 + ix) * 3 + c];
    }
  }
  const v16h* __restrict__ Af = (const v16h*)Wp;   // ktiles = 1
  v8f acc[4] = {};
#pragma unroll
  for (int mb = 0; mb < 4; ++mb)
    acc[mb] = __builtin_amdgcn_wmma_f32_16x16x32_f16(false, Af[(size_t)mb * 32 + lane],
                                                     false, bf, (short)0, acc[mb], false, false);
  const size_t base = (((size_t)bi * 256 + oy) * 256 + ox) * 64;   // NHWC, Cout=64
#pragma unroll
  for (int mb = 0; mb < 4; ++mb) {
    const int mbase = mb * 16 + half * 8;
    v8h o;
#pragma unroll
    for (int r = 0; r < 8; ++r) o[r] = (_Float16)fmaxf(acc[mb][r] + bias[mbase + r], 0.f);
    *(v8h*)(Y + base + mbase) = o;
  }
}

// ---------------- FPN (NHWC, C=128): dst(HxW) += up2(src(H/2 x W/2)) -------
__global__ void k_add_up2(_Float16* __restrict__ dst, const _Float16* __restrict__ src,
                          int B, int H, int W) {
  int i = blockIdx.x * blockDim.x + threadIdx.x;
  int tot = B * H * W * FPNC;
  if (i >= tot) return;
  int c = i & (FPNC - 1); int p = i >> 7;
  int x = p % W; int t = p / W; int y = t % H; int b = t / H;
  int hs = H >> 1, wsz = W >> 1;
  _Float16 s = src[((((size_t)b * hs + (y >> 1)) * wsz) + (x >> 1)) * FPNC + c];
  dst[i] = (_Float16)((float)dst[i] + (float)s);
}

// ---------------- decode (NHWC preds): scores / argmax / boxes -------------
__global__ void k_decode(const float* __restrict__ cls, const float* __restrict__ box,
                         const float* __restrict__ ctr,
                         float* __restrict__ scores, int* __restrict__ cids,
                         float* __restrict__ boxes,
                         int B, int H, int W, int sv, int base) {
  int i = blockIdx.x * blockDim.x + threadIdx.x;
  int hw = H * W;
  if (i >= B * hw) return;
  int b = i / hw; int r = i - b * hw; int y = r / W; int x = r - y * W;
  float pct = sigmoidf_(ctr[(size_t)b * hw + r]);
  const float* cp = cls + ((size_t)b * hw + r) * NCLS;   // contiguous classes
  float best = -1.f; int bc = 0;
  for (int c = 0; c < NCLS; ++c) {
    float s = sqrtf(sigmoidf_(cp[c]) * pct);
    if (s > best) { best = s; bc = c; }   // first max wins (argmax semantics)
  }
  const float* bp = box + ((size_t)b * hw + r) * 4;
  float fs = (float)sv;
  float xc = (x + 0.5f) * fs, yc = (y + 0.5f) * fs;
  float d0 = fmaxf(bp[0], 0.f) * fs, d1 = fmaxf(bp[1], 0.f) * fs;
  float d2 = fmaxf(bp[2], 0.f) * fs, d3 = fmaxf(bp[3], 0.f) * fs;
  float bx1 = fminf(fmaxf(xc - d0, 0.f), (float)IMG);
  float by1 = fminf(fmaxf(yc - d1, 0.f), (float)IMG);
  float bx2 = fminf(fmaxf(xc + d2, 0.f), (float)IMG);
  float by2 = fminf(fmaxf(yc + d3, 0.f), (float)IMG);
  int loc = base + r;
  scores[(size_t)b * NLOC + loc] = best;
  cids[(size_t)b * NLOC + loc]   = bc;
  float* bo = boxes + ((size_t)b * NLOC + loc) * 4;
  bo[0] = bx1; bo[1] = by1; bo[2] = bx2; bo[3] = by2;
}

// ---------------- per-image top-k via LDS bitonic sort ---------------------
// key = (~orderable(score) << 32) | idx  -> ascending sort == desc score, asc idx
__global__ __launch_bounds__(1024)
void k_topk(const float* __restrict__ scores, const int* __restrict__ cids,
            const float* __restrict__ boxes,
            float* __restrict__ out_topb, float* __restrict__ out_topc,
            float* __restrict__ top_s, int* __restrict__ top_cid) {
  __shared__ unsigned long long sk[NPAD];   // 64 KB LDS (WGP has 320 KB)
  int b = blockIdx.x, tid = threadIdx.x, T = blockDim.x;
  for (int i = tid; i < NPAD; i += T) {
    unsigned long long key;
    if (i < NLOC) {
      float s = scores[(size_t)b * NLOC + i];
      unsigned u = __float_as_uint(s);
      u = (u & 0x80000000u) ? ~u : (u | 0x80000000u);  // orderable ascending
      unsigned k32 = ~u;                               // descending score
      key = ((unsigned long long)k32 << 32) | (unsigned)i;
    } else {
      key = 0xFFFFFFFFFFFFFFFFull;                     // -inf pad -> sinks last
    }
    sk[i] = key;
  }
  for (unsigned k = 2; k <= (unsigned)NPAD; k <<= 1) {
    for (unsigned j = k >> 1; j > 0; j >>= 1) {
      __syncthreads();
      for (int i = tid; i < NPAD; i += T) {
        int ixj = i ^ (int)j;
        if (ixj > i) {
          bool up = ((i & (int)k) == 0);
          unsigned long long a = sk[i], c = sk[ixj];
          if ((a > c) == up) { sk[i] = c; sk[ixj] = a; }
        }
      }
    }
  }
  __syncthreads();
  for (int i = tid; i < TOPKN; i += T) {
    int idx = (int)(sk[i] & 0xFFFFFFFFull);
    float s = scores[(size_t)b * NLOC + idx];
    int   c = cids[(size_t)b * NLOC + idx];
    top_s[(size_t)b * TOPKN + i]   = s;
    top_cid[(size_t)b * TOPKN + i] = c;
    out_topc[(size_t)b * TOPKN + i] = (float)c;
    const float* bo = boxes + ((size_t)b * NLOC + idx) * 4;
    float* d = out_topb + ((size_t)b * TOPKN + i) * 4;
    d[0] = bo[0]; d[1] = bo[1]; d[2] = bo[2]; d[3] = bo[3];
  }
}

// ---------------- per-image greedy class-offset NMS in LDS -----------------
__global__ __launch_bounds__(256)
void k_nms(const float* __restrict__ topb, const float* __restrict__ tops,
           const int* __restrict__ topc,
           float* __restrict__ out_scores, float* __restrict__ out_keep) {
  __shared__ float X1[TOPKN], Y1[TOPKN], X2[TOPKN], Y2[TOPKN], AR[TOPKN], SC[TOPKN];
  __shared__ unsigned char KP[TOPKN];
  __shared__ float red[256];
  int b = blockIdx.x, tid = threadIdx.x, T = blockDim.x;
  const float* bb = topb + (size_t)b * TOPKN * 4;

  // max_c = max over all box coords of this image (reference semantics)
  float mx = -1e30f;
  for (int i = tid; i < TOPKN * 4; i += T) mx = fmaxf(mx, bb[i]);
  red[tid] = mx; __syncthreads();
  for (int s = 128; s > 0; s >>= 1) { if (tid < s) red[tid] = fmaxf(red[tid], red[tid + s]); __syncthreads(); }
  float maxc = red[0];

  for (int i = tid; i < TOPKN; i += T) {
    float off = (float)topc[(size_t)b * TOPKN + i] * (maxc + 1.f);
    float x1 = bb[i * 4 + 0] + off, y1 = bb[i * 4 + 1] + off;
    float x2 = bb[i * 4 + 2] + off, y2 = bb[i * 4 + 3] + off;
    X1[i] = x1; Y1[i] = y1; X2[i] = x2; Y2[i] = y2;
    AR[i] = (x2 - x1) * (y2 - y1);
    float s = tops[(size_t)b * TOPKN + i];
    SC[i] = s;
    KP[i] = (s > 0.3f) ? 1 : 0;
  }
  __syncthreads();

  for (int i = 0; i < TOPKN; ++i) {
    if (KP[i]) {
      float x1 = X1[i], y1 = Y1[i], x2 = X2[i], y2 = Y2[i], ai = AR[i];
      for (int j = i + 1 + tid; j < TOPKN; j += T) {
        float xx1 = fmaxf(x1, X1[j]), yy1 = fmaxf(y1, Y1[j]);
        float xx2 = fminf(x2, X2[j]), yy2 = fminf(y2, Y2[j]);
        float inter = fmaxf(xx2 - xx1, 0.f) * fmaxf(yy2 - yy1, 0.f);
        float iou = inter / (ai + AR[j] - inter + 1e-8f);
        if (iou > 0.5f) KP[j] = 0;
      }
    }
    __syncthreads();
  }

  for (int i = tid; i < TOPKN; i += T) {
    float k = KP[i] ? 1.f : 0.f;
    out_scores[(size_t)b * TOPKN + i] = SC[i] * k;
    out_keep[(size_t)b * TOPKN + i]   = k;
  }
}

// ===========================================================================
extern "C" void kernel_launch(void* const* d_in, const int* in_sizes, int n_in,
                              void* d_out, int out_size, void* d_ws, size_t ws_size,
                              hipStream_t stream) {
  (void)in_sizes; (void)n_in; (void)out_size; (void)ws_size;

  // Input order: jax-canonical (sorted dict keys) pytree flatten of
  // {'images', 'params':{'bb','box_stem','cls_stem','lat','out',
  //                      'pred_box','pred_cls','pred_ctr'}}
  const float* images = (const float*)d_in[0];
  const float *Wb[5], *Bb[5];
  for (int i = 0; i < 5; ++i) { Wb[i] = (const float*)d_in[1 + 2 * i]; Bb[i] = (const float*)d_in[2 + 2 * i]; }
  const float* Wbs[2] = {(const float*)d_in[11], (const float*)d_in[13]};
  const float* Bbs[2] = {(const float*)d_in[12], (const float*)d_in[14]};
  const float* Wcs[2] = {(const float*)d_in[15], (const float*)d_in[17]};
  const float* Bcs[2] = {(const float*)d_in[16], (const float*)d_in[18]};
  const float* Wlat[3] = {(const float*)d_in[19], (const float*)d_in[21], (const float*)d_in[23]};  // p3,p4,p5
  const float* Blat[3] = {(const float*)d_in[20], (const float*)d_in[22], (const float*)d_in[24]};
  const float* Wo[3] = {(const float*)d_in[25], (const float*)d_in[27], (const float*)d_in[29]};    // p3,p4,p5
  const float* Bo[3] = {(const float*)d_in[26], (const float*)d_in[28], (const float*)d_in[30]};
  const float* Wpb = (const float*)d_in[31]; const float* Bpb = (const float*)d_in[32];
  const float* Wpc = (const float*)d_in[33]; const float* Bpc = (const float*)d_in[34];
  const float* Wpt = (const float*)d_in[35]; const float* Bpt = (const float*)d_in[36];

  // ---- workspace bump allocator (total ~200 MB) ----
  char* wsb = (char*)d_ws;
  size_t off = 0;
  auto alloc = [&](size_t bytes) -> void* {
    void* p = wsb + off;
    off += (bytes + 255) & ~(size_t)255;
    return p;
  };
  auto a16 = [&](size_t n) -> _Float16* { return (_Float16*)alloc(n * 2); };
  auto af  = [&](size_t n) -> float*    { return (float*)alloc(n * 4); };
  auto ai  = [&](size_t n) -> int*      { return (int*)alloc(n * 4); };

  // All activations NHWC f16
  _Float16* img16 = a16((size_t)NB * 512 * 512 * 3);
  _Float16* x1 = a16((size_t)NB * 256 * 256 * 64);
  _Float16* x2 = a16((size_t)NB * 128 * 128 * 128);
  _Float16* x3 = a16((size_t)NB * 64 * 64 * 256);
  _Float16* x4 = a16((size_t)NB * 32 * 32 * 512);
  _Float16* x5 = a16((size_t)NB * 16 * 16 * 512);
  _Float16* l3 = a16((size_t)NB * 64 * 64 * FPNC);
  _Float16* l4 = a16((size_t)NB * 32 * 32 * FPNC);
  _Float16* l5 = a16((size_t)NB * 16 * 16 * FPNC);
  _Float16* p3 = a16((size_t)NB * 64 * 64 * FPNC);
  _Float16* p4 = a16((size_t)NB * 32 * 32 * FPNC);
  _Float16* p5 = a16((size_t)NB * 16 * 16 * FPNC);
  _Float16* t0 = a16((size_t)NB * 64 * 64 * FPNC);
  _Float16* t1 = a16((size_t)NB * 64 * 64 * FPNC);
  float* cls_[3] = { af((size_t)NB * 4096 * NCLS), af((size_t)NB * 1024 * NCLS), af((size_t)NB * 256 * NCLS) };
  float* box_[3] = { af((size_t)NB * 4096 * 4),    af((size_t)NB * 1024 * 4),    af((size_t)NB * 256 * 4) };
  float* ctr_[3] = { af((size_t)NB * 4096),        af((size_t)NB * 1024),        af((size_t)NB * 256) };
  float* scores = af((size_t)NB * NLOC);
  int*   cids   = ai((size_t)NB * NLOC);
  float* boxesd = af((size_t)NB * NLOC * 4);
  float* top_s  = af((size_t)NB * TOPKN);
  int*   top_c  = ai((size_t)NB * TOPKN);

  // packed weights, fragment order [Mp/16][Kp/32][lane][16] f16, tap-major K
  _Float16* pw_bb[5];
  const int bbCout[5] = {64, 128, 256, 512, 512};
  const int bbCin[5]  = {3, 64, 128, 256, 512};
  for (int i = 0; i < 5; ++i) {
    int K = bbCin[i] * 9, Kp = (K + 31) & ~31;
    pw_bb[i] = a16((size_t)bbCout[i] * Kp);
  }
  _Float16* pw_lat[3] = { a16((size_t)128 * 256), a16((size_t)128 * 512), a16((size_t)128 * 512) };
  _Float16* pw_out[3] = { a16((size_t)128 * 1152), a16((size_t)128 * 1152), a16((size_t)128 * 1152) };
  _Float16* pw_cs[2]  = { a16((size_t)128 * 1152), a16((size_t)128 * 1152) };
  _Float16* pw_bs[2]  = { a16((size_t)128 * 1152), a16((size_t)128 * 1152) };
  _Float16* pw_pc = a16((size_t)32 * 1152);
  _Float16* pw_pb = a16((size_t)16 * 1152);
  _Float16* pw_pt = a16((size_t)16 * 1152);

  auto pack = [&](const float* w, _Float16* dst, int Cout, int Cin, int ks, int mblk) {
    int K = Cin * ks * ks, Kp = (K + 31) & ~31;
    int gran = 16 * mblk;
    int Mp = (Cout + gran - 1) / gran * gran;
    int tot = Mp * Kp;
    k_pack_w<<<(tot + 255) / 256, 256, 0, stream>>>(w, dst, Cout, Cin, ks, K, Kp, Mp);
  };
  auto conv = [&](const _Float16* X, const _Float16* Wp, const float* bias, void* Y,
                  int Cin, int Hin, int Win, int Cout, int Hout, int Wout,
                  int ks, int stride, int pad, int mblk, bool relu, bool outf32) {
    int K = Cin * ks * ks;
    int cshift = __builtin_ctz((unsigned)(Cin >> 5));   // Cin/32 is a power of 2
    dim3 g((unsigned)((NB * Hout * Wout + 15) / 16),
           (unsigned)((Cout + 16 * mblk - 1) / (16 * mblk)));
    dim3 blk(32);
    if (mblk == 4 && ks == 3 && relu)
      k_conv_wmma<4, 3, true, false><<<g, blk, 0, stream>>>(X, Wp, bias, Y, NB, Cin, Hin, Win, Cout, Hout, Wout, stride, pad, K, cshift);
    else if (mblk == 4 && ks == 3)
      k_conv_wmma<4, 3, false, false><<<g, blk, 0, stream>>>(X, Wp, bias, Y, NB, Cin, Hin, Win, Cout, Hout, Wout, stride, pad, K, cshift);
    else if (mblk == 4 && ks == 1)
      k_conv_wmma<4, 1, false, false><<<g, blk, 0, stream>>>(X, Wp, bias, Y, NB, Cin, Hin, Win, Cout, Hout, Wout, stride, pad, K, cshift);
    else if (mblk == 2)
      k_conv_wmma<2, 3, false, true><<<g, blk, 0, stream>>>(X, Wp, bias, Y, NB, Cin, Hin, Win, Cout, Hout, Wout, stride, pad, K, cshift);
    else
      k_conv_wmma<1, 3, false, true><<<g, blk, 0, stream>>>(X, Wp, bias, Y, NB, Cin, Hin, Win, Cout, Hout, Wout, stride, pad, K, cshift);
  };

  // ---- convert input to NHWC f16, pack weights ----
  {
    int n = NB * 3 * 512 * 512;
    k_nchw_to_nhwc<<<(n + 255) / 256, 256, 0, stream>>>(images, img16, NB, 3, 512, 512);
  }
  for (int i = 0; i < 5; ++i) pack(Wb[i], pw_bb[i], bbCout[i], bbCin[i], 3, 4);
  pack(Wlat[0], pw_lat[0], 128, 256, 1, 4);
  pack(Wlat[1], pw_lat[1], 128, 512, 1, 4);
  pack(Wlat[2], pw_lat[2], 128, 512, 1, 4);
  for (int i = 0; i < 3; ++i) pack(Wo[i], pw_out[i], 128, 128, 3, 4);
  for (int i = 0; i < 2; ++i) { pack(Wcs[i], pw_cs[i], 128, 128, 3, 4); pack(Wbs[i], pw_bs[i], 128, 128, 3, 4); }
  pack(Wpc, pw_pc, NCLS, 128, 3, 2);
  pack(Wpb, pw_pb, 4, 128, 3, 1);
  pack(Wpt, pw_pt, 1, 128, 3, 1);

  // ---- backbone: 5 stride-2 3x3 convs + ReLU (SAME on even dims -> pad 0) ----
  {
    dim3 g((unsigned)((NB * 256 * 256) / 16)), blk(32);
    k_conv1_wmma<<<g, blk, 0, stream>>>(img16, pw_bb[0], Bb[0], x1);
  }
  conv(x1, pw_bb[1], Bb[1], x2,  64, 256, 256, 128, 128, 128, 3, 2, 0, 4, true, false);
  conv(x2, pw_bb[2], Bb[2], x3, 128, 128, 128, 256,  64,  64, 3, 2, 0, 4, true, false);
  conv(x3, pw_bb[3], Bb[3], x4, 256,  64,  64, 512,  32,  32, 3, 2, 0, 4, true, false);
  conv(x4, pw_bb[4], Bb[4], x5, 512,  32,  32, 512,  16,  16, 3, 2, 0, 4, true, false);

  // ---- FPN lateral (1x1) + top-down up2 adds ----
  conv(x5, pw_lat[2], Blat[2], l5, 512, 16, 16, 128, 16, 16, 1, 1, 0, 4, false, false);
  conv(x4, pw_lat[1], Blat[1], l4, 512, 32, 32, 128, 32, 32, 1, 1, 0, 4, false, false);
  { int n = NB * 32 * 32 * FPNC; k_add_up2<<<(n + 255) / 256, 256, 0, stream>>>(l4, l5, NB, 32, 32); }
  conv(x3, pw_lat[0], Blat[0], l3, 256, 64, 64, 128, 64, 64, 1, 1, 0, 4, false, false);
  { int n = NB * 64 * 64 * FPNC; k_add_up2<<<(n + 255) / 256, 256, 0, stream>>>(l3, l4, NB, 64, 64); }

  // ---- FPN output convs (3x3 s1 pad1, no relu) ----
  conv(l3, pw_out[0], Bo[0], p3, 128, 64, 64, 128, 64, 64, 3, 1, 1, 4, false, false);
  conv(l4, pw_out[1], Bo[1], p4, 128, 32, 32, 128, 32, 32, 3, 1, 1, 4, false, false);
  conv(l5, pw_out[2], Bo[2], p5, 128, 16, 16, 128, 16, 16, 3, 1, 1, 4, false, false);

  // ---- heads + decode per level ----
  const _Float16* plv[3] = { p3, p4, p5 };
  const int Hl[3] = {64, 32, 16}, Wl[3] = {64, 32, 16}, Sl[3] = {8, 16, 32}, Basel[3] = {0, 4096, 5120};
  for (int L = 0; L < 3; ++L) {
    int H = Hl[L], W = Wl[L];
    // cls branch
    conv(plv[L], pw_cs[0], Bcs[0], t0, 128, H, W, 128, H, W, 3, 1, 1, 4, true, false);
    conv(t0,     pw_cs[1], Bcs[1], t1, 128, H, W, 128, H, W, 3, 1, 1, 4, true, false);
    conv(t1,     pw_pc,    Bpc, cls_[L], 128, H, W, NCLS, H, W, 3, 1, 1, 2, false, true);
    // box branch
    conv(plv[L], pw_bs[0], Bbs[0], t0, 128, H, W, 128, H, W, 3, 1, 1, 4, true, false);
    conv(t0,     pw_bs[1], Bbs[1], t1, 128, H, W, 128, H, W, 3, 1, 1, 4, true, false);
    conv(t1,     pw_pb,    Bpb, box_[L], 128, H, W, 4, H, W, 3, 1, 1, 1, false, true);
    conv(t1,     pw_pt,    Bpt, ctr_[L], 128, H, W, 1, H, W, 3, 1, 1, 1, false, true);
    int n = NB * H * W;
    k_decode<<<(n + 255) / 256, 256, 0, stream>>>(cls_[L], box_[L], ctr_[L],
                                                  scores, cids, boxesd,
                                                  NB, H, W, Sl[L], Basel[L]);
  }

  // ---- top-k + NMS -> d_out = [topb(32000) | final_scores(8000) | topc(8000) | keep(8000)] ----
  float* out_topb   = (float*)d_out;
  float* out_scores = (float*)d_out + (size_t)NB * TOPKN * 4;
  float* out_topc   = out_scores + (size_t)NB * TOPKN;
  float* out_keep   = out_topc + (size_t)NB * TOPKN;

  k_topk<<<NB, 1024, 0, stream>>>(scores, cids, boxesd, out_topb, out_topc, top_s, top_c);
  k_nms<<<NB, 256, 0, stream>>>(out_topb, top_s, top_c, out_scores, out_keep);
}
